// HierAttn_42545946034811
// MI455X (gfx1250) — matrix operand order repaired and checked by reading, not compiled, not run on hardware
//
#include <hip/hip_runtime.h>
#include <hip/hip_bf16.h>

typedef __attribute__((ext_vector_type(16))) __bf16 v16bf;
typedef __attribute__((ext_vector_type(8)))  __bf16 v8bf;
typedef __attribute__((ext_vector_type(8)))  float  v8f;
typedef __attribute__((ext_vector_type(4)))  unsigned int u32x4;
typedef __attribute__((ext_vector_type(8)))  int i32x8;
typedef __attribute__((ext_vector_type(4)))  int i32x4;

union BF16x16 { v16bf v; v8bf h[2]; };
union U4x8    { uint4 u; __bf16 hh[8]; };

static constexpr int Bc = 8, Tc = 4096, Ec = 1024, Hc = 16, DHc = 64;
static constexpr int W0c = 128, NL1 = 16, NL2 = 8;

__device__ __forceinline__ v8f wmma_bf16(const BF16x16& a, const BF16x16& b, v8f c) {
  return __builtin_amdgcn_wmma_f32_16x16x32_bf16(false, a.v, false, b.v, (short)0, c,
                                                 false, false);
}

#if __has_builtin(__builtin_amdgcn_tensor_load_to_lds) && \
    __has_builtin(__builtin_amdgcn_s_wait_tensorcnt)
#define USE_TDM 1
#endif

#ifdef USE_TDM
// ---------------------------------------------------------------------------
// Tensor Data Mover: 2D tile (tile1 rows x tile0 elems, 2-byte elements) from
// a row-major [tensor_rows x tdim0] bf16 tensor into LDS at lds_byte_off.
// D# bitfields per CDNA5 ISA ch.8 (group0: count/lds/global/type; group1:
// data_size, dims, tile dims, stride).  EXEC-independent; tracked by TENSORcnt.
// This toolchain uses the 6-arg builtin: (g0, g1, g2, g3, g4, cpol).
// ---------------------------------------------------------------------------
__device__ __forceinline__ void tdm_load_tile(const __bf16* gbase, unsigned lds_byte_off,
                                              unsigned tile1_rows, unsigned tile0_elems,
                                              unsigned tdim0, unsigned tensor_rows) {
  unsigned long long ga = (unsigned long long)(const void*)gbase;
  u32x4 g0;
  g0.x = 1u;                                                     // count=1 (user D#)
  g0.y = lds_byte_off;                                           // lds_addr
  g0.z = (unsigned)(ga & 0xFFFFFFFFu);                           // global_addr lo
  g0.w = (unsigned)((ga >> 32) & 0x01FFFFFFu) | (2u << 30);      // addr hi | type=2
  i32x8 g1;
  g1[0] = 1 << 16;                                               // data_size=1 -> 2B
  g1[1] = (int)((tdim0 & 0xFFFFu) << 16);                        // tensor_dim0 lo16
  g1[2] = (int)(((tdim0 >> 16) & 0xFFFFu) |
                ((tensor_rows & 0xFFFFu) << 16));                // dim0 hi | dim1 lo
  g1[3] = (int)(((tensor_rows >> 16) & 0xFFFFu) |
                ((tile0_elems & 0xFFFFu) << 16));                // dim1 hi | tile_dim0
  g1[4] = (int)(tile1_rows & 0xFFFFu);                           // tile_dim1 (dim2=0)
  g1[5] = (int)tdim0;                                            // dim0_stride lo32
  g1[6] = 0;                                                     // stride hi | d1s lo
  g1[7] = 0;
  i32x4 z4 = {0, 0, 0, 0};
  i32x8 z8 = {0, 0, 0, 0, 0, 0, 0, 0};
  __builtin_amdgcn_tensor_load_to_lds(g0, g1, z4, z4, z8, 0);
}
#endif

// ---------------------------------------------------------------------------
// x (f32) -> bf16, vectorized 4-wide
// ---------------------------------------------------------------------------
__global__ __launch_bounds__(256) void cvt_kernel(const float* __restrict__ x,
                                                  __bf16* __restrict__ xb, size_t n4) {
  size_t i = (size_t)blockIdx.x * blockDim.x + threadIdx.x;
  if (i >= n4) return;
  float4 f = ((const float4*)x)[i];
  union { __bf16 h[4]; uint2 u; } r;
  r.h[0] = (__bf16)f.x; r.h[1] = (__bf16)f.y; r.h[2] = (__bf16)f.z; r.h[3] = (__bf16)f.w;
  ((uint2*)xb)[i] = r.u;
}

// ---------------------------------------------------------------------------
// W_eff = W + 2 * A @ B  (rank-8 LoRA fold), cast to bf16.  A==nullptr: cast only.
// ---------------------------------------------------------------------------
__global__ __launch_bounds__(256) void fold_kernel(const float* __restrict__ W,
                                                   const float* __restrict__ A,
                                                   const float* __restrict__ Bm,
                                                   __bf16* __restrict__ out) {
  int idx = blockIdx.x * blockDim.x + threadIdx.x;
  if (idx >= Ec * Ec) return;
  int o = idx >> 10, i = idx & 1023;
  float wv = W[idx];
  if (A != nullptr) {
    float acc = 0.f;
#pragma unroll
    for (int r = 0; r < 8; r++) acc += A[o * 8 + r] * Bm[r * Ec + i];
    wv += 2.0f * acc;
  }
  out[idx] = (__bf16)wv;
}

// ---------------------------------------------------------------------------
// TN GEMM: C[m,n] = sum_k A[m,k]*B[n,k].  A: MxK bf16 rm, B: NxK bf16 rm.
// 128x128 block tile, BK=64, double-buffered LDS filled by TDM (8 waves each
// DMA one 32-row slice), 8 waves in 4(M)x2(N): 32x64 per wave = 8 WMMA/K32.
// ---------------------------------------------------------------------------
template <bool OUT_F32>
__global__ __launch_bounds__(256) void gemm_tn(const __bf16* __restrict__ A,
                                               const __bf16* __restrict__ Bw,
                                               void* __restrict__ Cout,
                                               int M, int N, int K) {
  constexpr int BM = 128, BN = 128, BK = 64;
  __shared__ __align__(16) __bf16 As[2][BM * BK];
  __shared__ __align__(16) __bf16 Bs[2][BN * BK];

  const int t = threadIdx.x;
  const int wv = t >> 5, lane = t & 31, ln = lane & 15;
  const bool hi = lane >= 16;
  const int wm = wv >> 1, wn = wv & 1;  // 4 x 2 wave grid
  const int m0 = blockIdx.y * BM, n0 = blockIdx.x * BN;
  const int NK = K / BK;

  v8f acc[2][4] = {};

#ifdef USE_TDM
  // Branchless per-wave slice select (TDM ignores EXEC: no divergent ifs).
  const int isB = wv >> 2;          // waves 0-3: A slices, 4-7: B slices
  const int slice = wv & 3;
  const __bf16* gsl = (isB ? Bw : A) + (size_t)((isB ? n0 : m0) + slice * 32) * K;
  const unsigned ldsA0 = (unsigned)(unsigned long long)&As[0][slice * 32 * BK];
  const unsigned ldsB0 = (unsigned)(unsigned long long)&Bs[0][slice * 32 * BK];
  const unsigned ldsA1 = (unsigned)(unsigned long long)&As[1][slice * 32 * BK];
  const unsigned ldsB1 = (unsigned)(unsigned long long)&Bs[1][slice * 32 * BK];
  const unsigned lds0 = isB ? ldsB0 : ldsA0;
  const unsigned lds1 = isB ? ldsB1 : ldsA1;
  const unsigned trows = (unsigned)(isB ? N : M);

  tdm_load_tile(gsl, lds0, 32, BK, (unsigned)K, trows);  // prologue: tile 0
#endif

  for (int kt = 0; kt < NK; ++kt) {
    const int cur = kt & 1;
#ifdef USE_TDM
    __syncthreads();  // everyone done reading buf[cur^1] (tile kt-1)
    if (kt + 1 < NK) {
      tdm_load_tile(gsl + (size_t)(kt + 1) * BK, cur ? lds0 : lds1, 32, BK,
                    (unsigned)K, trows);
      __builtin_amdgcn_s_wait_tensorcnt(1);  // tile kt landed; kt+1 in flight
    } else {
      __builtin_amdgcn_s_wait_tensorcnt(0);
    }
    __syncthreads();  // tile kt visible to all waves
#else
    __syncthreads();
#pragma unroll
    for (int i = 0; i < 4; i++) {  // A tile: 1024 chunks of 8 halves
      int c = t + 256 * i, row = c >> 3, kc = (c & 7) * 8;
      *(uint4*)&As[cur][row * BK + kc] =
          *(const uint4*)&A[(size_t)(m0 + row) * K + kt * BK + kc];
    }
#pragma unroll
    for (int i = 0; i < 4; i++) {  // B tile
      int c = t + 256 * i, row = c >> 3, kc = (c & 7) * 8;
      *(uint4*)&Bs[cur][row * BK + kc] =
          *(const uint4*)&Bw[(size_t)(n0 + row) * K + kt * BK + kc];
    }
    __syncthreads();
#endif

#pragma unroll
    for (int ks = 0; ks < 2; ++ks) {  // two K=32 steps per BK=64 tile
      BF16x16 af[2];
#pragma unroll
      for (int mi = 0; mi < 2; mi++) {
        const int arow = wm * 32 + mi * 16 + ln;
        const int akb = (hi ? 8 : 0) + 32 * ks;
        af[mi].h[0] = *(const v8bf*)&As[cur][arow * BK + akb];
        af[mi].h[1] = *(const v8bf*)&As[cur][arow * BK + akb + 16];
      }
#pragma unroll
      for (int j = 0; j < 4; j++) {
        BF16x16 bf_;
        const int brow = wn * 64 + j * 16 + ln;
        const int bkb = (hi ? 8 : 0) + 32 * ks;
        bf_.h[0] = *(const v8bf*)&Bs[cur][brow * BK + bkb];
        bf_.h[1] = *(const v8bf*)&Bs[cur][brow * BK + bkb + 16];
#pragma unroll
        for (int mi = 0; mi < 2; mi++) acc[mi][j] = wmma_bf16(af[mi], bf_, acc[mi][j]);
      }
    }
  }

  const int rb = m0 + wm * 32 + (hi ? 8 : 0);
#pragma unroll
  for (int mi = 0; mi < 2; mi++)
#pragma unroll
    for (int j = 0; j < 4; j++) {
      const int col = n0 + wn * 64 + j * 16 + ln;
#pragma unroll
      for (int v = 0; v < 8; v++) {
        const size_t off = (size_t)(rb + mi * 16 + v) * N + col;
        if (OUT_F32) ((float*)Cout)[off] = acc[mi][j][v];
        else         ((__bf16*)Cout)[off] = (__bf16)acc[mi][j][v];
      }
    }
}

// ---------------------------------------------------------------------------
// Hierarchical pooling: K1/V1 = mean of 8-token chunks (last 16 chunks),
// K2/V2 = mean of 16-token pairs (last 8).  Tokens 3840..3967.
// ---------------------------------------------------------------------------
__global__ __launch_bounds__(256) void pool_kernel(const __bf16* __restrict__ Kb,
                                                   const __bf16* __restrict__ Vb,
                                                   __bf16* __restrict__ K1, __bf16* __restrict__ V1,
                                                   __bf16* __restrict__ K2, __bf16* __restrict__ V2) {
  int idx = blockIdx.x * blockDim.x + threadIdx.x;  // Bc*24*Ec threads
  int e = idx & 1023;
  int r = idx >> 10;
  int j = r % 24, b = r / 24;
  if (b >= Bc) return;
  if (j < NL1) {
    size_t base = ((size_t)b * Tc + 3840 + 8 * j) * Ec + e;
    float sk = 0.f, sv = 0.f;
#pragma unroll
    for (int i = 0; i < 8; i++) {
      sk += (float)Kb[base + (size_t)i * Ec];
      sv += (float)Vb[base + (size_t)i * Ec];
    }
    K1[((size_t)b * NL1 + j) * Ec + e] = (__bf16)(sk * 0.125f);
    V1[((size_t)b * NL1 + j) * Ec + e] = (__bf16)(sv * 0.125f);
  } else {
    int p = j - NL1;
    size_t base = ((size_t)b * Tc + 3840 + 16 * p) * Ec + e;
    float sk = 0.f, sv = 0.f;
#pragma unroll
    for (int i = 0; i < 16; i++) {
      sk += (float)Kb[base + (size_t)i * Ec];
      sv += (float)Vb[base + (size_t)i * Ec];
    }
    K2[((size_t)b * NL2 + p) * Ec + e] = (__bf16)(sk * 0.0625f);
    V2[((size_t)b * NL2 + p) * Ec + e] = (__bf16)(sv * 0.0625f);
  }
}

// ---------------------------------------------------------------------------
// Fused 3-level attention.  Block: 128 thr (4 waves), 64 query rows, one (b,h).
// WMMA QK^T -> f32 softmax (wave shuffles) -> P via per-wave LDS -> WMMA PV.
// Levels weighted by softmax(lvl_w), accumulated into one f32 accumulator.
// ---------------------------------------------------------------------------
__global__ __launch_bounds__(128) void attn_kernel(
    const __bf16* __restrict__ Qb, const __bf16* __restrict__ Kb, const __bf16* __restrict__ Vb,
    const __bf16* __restrict__ K1, const __bf16* __restrict__ V1,
    const __bf16* __restrict__ K2, const __bf16* __restrict__ V2,
    const float* __restrict__ lvlw, __bf16* __restrict__ outb) {
  __shared__ __align__(16) __bf16 Ks0[W0c * DHc];   // [key][d]
  __shared__ __align__(16) __bf16 Vs0[DHc * W0c];   // [d][key] (transposed)
  __shared__ __align__(16) __bf16 Ks1[NL1 * DHc];
  __shared__ __align__(16) __bf16 Vs1[DHc * 32];    // keys padded to 32 (zeros)
  __shared__ __align__(16) __bf16 Ks2[16 * DHc];    // keys padded to 16
  __shared__ __align__(16) __bf16 Vs2[DHc * 32];
  __shared__ __align__(16) __bf16 Ps[4 * 16 * W0c]; // per-wave P scratch

  const int t = threadIdx.x;
  const int wv = t >> 5, lane = t & 31, ln = lane & 15;
  const bool hi = lane >= 16;
  const int bh = blockIdx.y, b = bh / Hc, h = bh % Hc;
  const int trow0 = blockIdx.x * 64;

  const __bf16* Khead = Kb + ((size_t)b * Tc + (Tc - W0c)) * Ec + h * DHc;
  const __bf16* Vhead = Vb + ((size_t)b * Tc + (Tc - W0c)) * Ec + h * DHc;

#pragma unroll
  for (int i = 0; i < 8; i++) {
    int c = t + 128 * i, key = c >> 3, dc = (c & 7) * 8;
    *(uint4*)&Ks0[key * DHc + dc] = *(const uint4*)&Khead[(size_t)key * Ec + dc];
  }
#pragma unroll
  for (int i = 0; i < 8; i++) {
    int c = t + 128 * i, key = c >> 3, dc = (c & 7) * 8;
    U4x8 u; u.u = *(const uint4*)&Vhead[(size_t)key * Ec + dc];
#pragma unroll
    for (int q = 0; q < 8; q++) Vs0[(dc + q) * W0c + key] = u.hh[q];
  }
  {
    int key = t >> 3, dc = (t & 7) * 8;
    *(uint4*)&Ks1[key * DHc + dc] =
        *(const uint4*)&K1[((size_t)b * NL1 + key) * Ec + h * DHc + dc];
    U4x8 u; u.u = *(const uint4*)&V1[((size_t)b * NL1 + key) * Ec + h * DHc + dc];
#pragma unroll
    for (int q = 0; q < 8; q++) Vs1[(dc + q) * 32 + key] = u.hh[q];
  }
#pragma unroll
  for (int i = 0; i < 8; i++) {  // Vs1 keys 16..31 zero (64*16 elems)
    int idx = t * 8 + i, d = idx >> 4, kk = idx & 15;
    Vs1[d * 32 + 16 + kk] = (__bf16)0.f;
  }
  if (t < 64) {
    int key = t >> 3, dc = (t & 7) * 8;
    *(uint4*)&Ks2[key * DHc + dc] =
        *(const uint4*)&K2[((size_t)b * NL2 + key) * Ec + h * DHc + dc];
    U4x8 u; u.u = *(const uint4*)&V2[((size_t)b * NL2 + key) * Ec + h * DHc + dc];
#pragma unroll
    for (int q = 0; q < 8; q++) Vs2[(dc + q) * 32 + key] = u.hh[q];
  } else {
    int c = t - 64, key = 8 + (c >> 3), dc = (c & 7) * 8;
    uint4 z = {0u, 0u, 0u, 0u};
    *(uint4*)&Ks2[key * DHc + dc] = z;
  }
#pragma unroll
  for (int i = 0; i < 12; i++) {  // Vs2 keys 8..31 zero (64*24 elems)
    int idx = t * 12 + i, d = idx / 24, kk = idx % 24;
    Vs2[d * 32 + 8 + kk] = (__bf16)0.f;
  }
  __syncthreads();

  float l0 = lvlw[0], l1 = lvlw[1], l2 = lvlw[2];
  float mw = fmaxf(l0, fmaxf(l1, l2));
  float e0 = expf(l0 - mw), e1 = expf(l1 - mw), e2 = expf(l2 - mw);
  float winv = 1.f / (e0 + e1 + e2);
  const float wl0 = e0 * winv, wl1 = e1 * winv, wl2 = e2 * winv;
  const float scale = 0.125f;  // 1/sqrt(64)

  const __bf16* Qhead = Qb + ((size_t)b * Tc + trow0 + wv * 16 + ln) * Ec + h * DHc;
  BF16x16 qf[2];
#pragma unroll
  for (int kk = 0; kk < 2; kk++) {
    int akb = (hi ? 8 : 0) + 32 * kk;
    qf[kk].h[0] = *(const v8bf*)&Qhead[akb];
    qf[kk].h[1] = *(const v8bf*)&Qhead[akb + 16];
  }
  __bf16* Psw = &Ps[wv * 16 * W0c];
  v8f o[4] = {};

  // ================= level 0: 128 keys =================
  {
    v8f s[8] = {};
#pragma unroll
    for (int kk = 0; kk < 2; kk++)
#pragma unroll
      for (int j = 0; j < 8; j++) {
        BF16x16 kf;
        int keyrow = j * 16 + ln, kb = (hi ? 8 : 0) + 32 * kk;
        kf.h[0] = *(const v8bf*)&Ks0[keyrow * DHc + kb];
        kf.h[1] = *(const v8bf*)&Ks0[keyrow * DHc + kb + 16];
        s[j] = wmma_bf16(qf[kk], kf, s[j]);
      }
#pragma unroll
    for (int v = 0; v < 8; v++) {
      float mv = s[0][v];
#pragma unroll
      for (int j = 1; j < 8; j++) mv = fmaxf(mv, s[j][v]);
      for (int off = 1; off < 16; off <<= 1) mv = fmaxf(mv, __shfl_xor(mv, off, 32));
      float sum = 0.f;
#pragma unroll
      for (int j = 0; j < 8; j++) { float p = expf(scale * (s[j][v] - mv)); s[j][v] = p; sum += p; }
      for (int off = 1; off < 16; off <<= 1) sum += __shfl_xor(sum, off, 32);
      float f = wl0 / sum;
      int prow = v + (hi ? 8 : 0);
#pragma unroll
      for (int j = 0; j < 8; j++) Psw[prow * W0c + j * 16 + ln] = (__bf16)(s[j][v] * f);
    }
    __syncthreads();
#pragma unroll
    for (int kk = 0; kk < 4; kk++) {
      BF16x16 pf;
      int akb = (hi ? 8 : 0) + 32 * kk;
      pf.h[0] = *(const v8bf*)&Psw[ln * W0c + akb];
      pf.h[1] = *(const v8bf*)&Psw[ln * W0c + akb + 16];
#pragma unroll
      for (int jd = 0; jd < 4; jd++) {
        BF16x16 vf;
        int drow = jd * 16 + ln, kb = (hi ? 8 : 0) + 32 * kk;
        vf.h[0] = *(const v8bf*)&Vs0[drow * W0c + kb];
        vf.h[1] = *(const v8bf*)&Vs0[drow * W0c + kb + 16];
        o[jd] = wmma_bf16(pf, vf, o[jd]);
      }
    }
    __syncthreads();
  }

  // ================= levels 1 & 2: 16-key tiles (lvl2 masked to 8) ==========
#pragma unroll
  for (int lvl = 1; lvl <= 2; lvl++) {
    const __bf16* Ksl = (lvl == 1) ? Ks1 : Ks2;
    const __bf16* Vsl = (lvl == 1) ? Vs1 : Vs2;
    const float wl = (lvl == 1) ? wl1 : wl2;
    v8f s1 = {};
#pragma unroll
    for (int kk = 0; kk < 2; kk++) {
      BF16x16 kf;
      int kb = (hi ? 8 : 0) + 32 * kk;
      kf.h[0] = *(const v8bf*)&Ksl[ln * DHc + kb];
      kf.h[1] = *(const v8bf*)&Ksl[ln * DHc + kb + 16];
      s1 = wmma_bf16(qf[kk], kf, s1);
    }
#pragma unroll
    for (int v = 0; v < 8; v++) {
      float sv = s1[v];
      if (lvl == 2 && ln >= NL2) sv = -INFINITY;  // mask padded keys
      float mv = sv;
      for (int off = 1; off < 16; off <<= 1) mv = fmaxf(mv, __shfl_xor(mv, off, 32));
      float p = expf(scale * (sv - mv));
      float sum = p;
      for (int off = 1; off < 16; off <<= 1) sum += __shfl_xor(sum, off, 32);
      int prow = v + (hi ? 8 : 0);
      Psw[prow * W0c + ln] = (__bf16)(p * wl / sum);
      Psw[prow * W0c + 16 + ln] = (__bf16)0.f;  // zero-pad K 16..31
    }
    __syncthreads();
    {
      BF16x16 pf;
      int akb = hi ? 8 : 0;
      pf.h[0] = *(const v8bf*)&Psw[ln * W0c + akb];
      pf.h[1] = *(const v8bf*)&Psw[ln * W0c + akb + 16];
#pragma unroll
      for (int jd = 0; jd < 4; jd++) {
        BF16x16 vf;
        int drow = jd * 16 + ln, kb = hi ? 8 : 0;
        vf.h[0] = *(const v8bf*)&Vsl[drow * 32 + kb];
        vf.h[1] = *(const v8bf*)&Vsl[drow * 32 + kb + 16];
        o[jd] = wmma_bf16(pf, vf, o[jd]);
      }
    }
    __syncthreads();
  }

  __bf16* outp = outb + ((size_t)b * Tc + trow0 + wv * 16) * Ec + h * DHc;
#pragma unroll
  for (int jd = 0; jd < 4; jd++)
#pragma unroll
    for (int v = 0; v < 8; v++) {
      int r = v + (hi ? 8 : 0);
      outp[(size_t)r * Ec + jd * 16 + ln] = (__bf16)o[jd][v];
    }
}

__global__ void recon_kernel(float* __restrict__ out) {
  out[(size_t)Bc * Tc * Ec] = 0.f;  // recon is identically zero
}

// ---------------------------------------------------------------------------
extern "C" void kernel_launch(void* const* d_in, const int* in_sizes, int n_in,
                              void* d_out, int out_size, void* d_ws, size_t ws_size,
                              hipStream_t stream) {
  const float* x    = (const float*)d_in[0];
  const float* Wq_w = (const float*)d_in[1];
  const float* Wq_A = (const float*)d_in[2];
  const float* Wq_B = (const float*)d_in[3];
  const float* Wk_w = (const float*)d_in[4];
  const float* Wk_A = (const float*)d_in[5];
  const float* Wk_B = (const float*)d_in[6];
  const float* Wv_w = (const float*)d_in[7];
  const float* Wv_A = (const float*)d_in[8];
  const float* Wv_B = (const float*)d_in[9];
  const float* Wp   = (const float*)d_in[10];
  const float* lvlw = (const float*)d_in[11];

  const size_t BTE = (size_t)Bc * Tc * Ec;   // 33554432
  const size_t EE  = (size_t)Ec * Ec;        // 1048576

  char* ws = (char*)d_ws;
  size_t off = 0;
  __bf16* xb = (__bf16*)(ws + off); off += 2 * BTE;  // reused as attn out
  __bf16* qb = (__bf16*)(ws + off); off += 2 * BTE;
  __bf16* kb = (__bf16*)(ws + off); off += 2 * BTE;
  __bf16* vb = (__bf16*)(ws + off); off += 2 * BTE;
  __bf16* wq = (__bf16*)(ws + off); off += 2 * EE;
  __bf16* wk = (__bf16*)(ws + off); off += 2 * EE;
  __bf16* wvv= (__bf16*)(ws + off); off += 2 * EE;
  __bf16* wp = (__bf16*)(ws + off); off += 2 * EE;
  __bf16* k1 = (__bf16*)(ws + off); off += 2 * (size_t)Bc * NL1 * Ec;
  __bf16* v1 = (__bf16*)(ws + off); off += 2 * (size_t)Bc * NL1 * Ec;
  __bf16* k2 = (__bf16*)(ws + off); off += 2 * (size_t)Bc * NL2 * Ec;
  __bf16* v2 = (__bf16*)(ws + off); off += 2 * (size_t)Bc * NL2 * Ec;

  cvt_kernel<<<(BTE / 4 + 255) / 256, 256, 0, stream>>>(x, xb, BTE / 4);
  fold_kernel<<<(EE + 255) / 256, 256, 0, stream>>>(Wq_w, Wq_A, Wq_B, wq);
  fold_kernel<<<(EE + 255) / 256, 256, 0, stream>>>(Wk_w, Wk_A, Wk_B, wk);
  fold_kernel<<<(EE + 255) / 256, 256, 0, stream>>>(Wv_w, Wv_A, Wv_B, wvv);
  fold_kernel<<<(EE + 255) / 256, 256, 0, stream>>>(Wp, nullptr, nullptr, wp);

  dim3 gg(Ec / 128, (Bc * Tc) / 128);
  gemm_tn<false><<<gg, 256, 0, stream>>>(xb, wq, qb, Bc * Tc, Ec, Ec);
  gemm_tn<false><<<gg, 256, 0, stream>>>(xb, wk, kb, Bc * Tc, Ec, Ec);
  gemm_tn<false><<<gg, 256, 0, stream>>>(xb, wvv, vb, Bc * Tc, Ec, Ec);

  pool_kernel<<<(Bc * 24 * Ec) / 256, 256, 0, stream>>>(kb, vb, k1, v1, k2, v2);
  attn_kernel<<<dim3(Tc / 64, Bc * Hc), 128, 0, stream>>>(qb, kb, vb, k1, v1, k2, v2,
                                                          lvlw, xb);
  gemm_tn<true><<<gg, 256, 0, stream>>>(xb, wp, d_out, Bc * Tc, Ec, Ec);
  recon_kernel<<<1, 1, 0, stream>>>((float*)d_out);
}
